// GRUEmbeddingModel_77343771066503
// MI455X (gfx1250) — matrix operand order repaired and checked by reading, not compiled, not run on hardware
//
#include <hip/hip_runtime.h>
#include <hip/hip_bf16.h>
#include <math.h>

// ---------------------------------------------------------------------------
// GRUEmbeddingModel for MI455X (gfx1250, wave32).
//   * Layer-0/1 input projections: bf16 WMMA GEMM (v_wmma_f32_16x16x32_bf16),
//     f32 accumulation. W^T pre-packed into the wave32 B-operand layout.
//   * Recurrences: one persistent 128-thread block per direction, W_hh rows in
//     VGPRs, h + gates in LDS, split-barrier sync per step.
//   * Heads: trivial VALU kernel.
// ---------------------------------------------------------------------------

typedef __attribute__((ext_vector_type(16))) __bf16 v16bf;
typedef __attribute__((ext_vector_type(8)))  __bf16 v8bf;
typedef __attribute__((ext_vector_type(8)))  float  v8f;

#define N_STEPS 8192
#define HID     40
#define G3      120     // 3*HID gates per direction
#define NCOLS   240     // both directions' gates, stacked
#define NT      15      // N tiles = 240/16
#define K0      3077    // 3072 + 5
#define K0PAD   3104    // padded to multiple of 32
#define K0ITERS 97      // 3104/32
#define K1      80      // 2*HID
#define K1PAD   96
#define K1ITERS 3

// ---- workspace layout (bytes) ---------------------------------------------
static constexpr size_t OFF_XCAT = 0;                                    // bf16 [8192][3104]
static constexpr size_t SZ_XCAT  = (size_t)N_STEPS * K0PAD * 2;
static constexpr size_t OFF_BP0  = OFF_XCAT + SZ_XCAT;                   // bf16 packed W0^T
static constexpr size_t SZ_BP0   = (size_t)NT * K0ITERS * 512 * 2;
static constexpr size_t OFF_BP1  = OFF_BP0 + SZ_BP0;                     // bf16 packed W1^T
static constexpr size_t SZ_BP1   = (size_t)NT * K1ITERS * 512 * 2;
static constexpr size_t OFF_GI0  = OFF_BP1 + SZ_BP1;                     // f32 [8192][240]
static constexpr size_t SZ_GI    = (size_t)N_STEPS * NCOLS * 4;
static constexpr size_t OFF_H0BF = OFF_GI0 + SZ_GI;                      // bf16 [8192][96]
static constexpr size_t SZ_H0BF  = (size_t)N_STEPS * K1PAD * 2;
static constexpr size_t OFF_GI1  = OFF_H0BF + SZ_H0BF;                   // f32 [8192][240]
static constexpr size_t OFF_OUT1 = OFF_GI1 + SZ_GI;                      // f32 [8192][80]
static constexpr size_t SZ_OUT1  = (size_t)N_STEPS * 2 * HID * 4;
static constexpr size_t WS_NEED  = OFF_OUT1 + SZ_OUT1;                   // ~72.3 MB

// ---------------------------------------------------------------------------
// Build X_cat = concat(x, y) as bf16, padded to K0PAD columns.
__global__ void build_xcat(const float* __restrict__ x, const float* __restrict__ y,
                           __bf16* __restrict__ xcat) {
  size_t idx = (size_t)blockIdx.x * blockDim.x + threadIdx.x;
  if (idx >= (size_t)N_STEPS * K0PAD) return;
  int row = (int)(idx / K0PAD);
  int col = (int)(idx % K0PAD);
  float v = 0.f;
  if (col < 3072)      v = x[(size_t)row * 3072 + col];
  else if (col < K0)   v = y[(size_t)row * 5 + (col - 3072)];
  xcat[idx] = (__bf16)v;
}

__global__ void zero_bf16(__bf16* __restrict__ p, unsigned n) {
  unsigned i = blockIdx.x * blockDim.x + threadIdx.x;
  if (i < n) p[i] = (__bf16)0.f;
}

// ---------------------------------------------------------------------------
// Pack W (rows 0..119 = fwd gates, 120..239 = bwd gates; each row length K) into
// the wave32 WMMA B-operand layout: per (ntile, kiter), lane l holds column
// n = ntile*16 + (l&15), K = kiter*32 + (l>>4)*8 + (i&7) + (i>>3)*16 for
// i = 0..15 (32-byte chunk per lane).
__global__ void pack_w(const float* __restrict__ wf, const float* __restrict__ wb,
                       int K, int kiters, __bf16* __restrict__ out) {
  int idx = blockIdx.x * blockDim.x + threadIdx.x;
  int total = NT * kiters * 512;
  if (idx >= total) return;
  int i     = idx & 15;
  int lane  = (idx >> 4) & 31;
  int kiter = (idx >> 9) % kiters;
  int ntile = idx / (kiters << 9);
  int k = kiter * 32 + ((lane >> 4) << 3) + (i & 7) + ((i >> 3) << 4);
  int n = ntile * 16 + (lane & 15);
  float v = 0.f;
  if (k < K) v = (n < G3) ? wf[(size_t)n * K + k] : wb[(size_t)(n - G3) * K + k];
  out[idx] = (__bf16)v;
}

// ---------------------------------------------------------------------------
// C[M][240] = A[M][lda(bf16)] @ W^T via v_wmma_f32_16x16x32_bf16.
// grid = (M/16, 4), block = 128 (4 waves); wave w handles n-tile blockIdx.y*4+w.
__global__ __launch_bounds__(128) void gemm_bf16_wmma(
    const __bf16* __restrict__ A, const __bf16* __restrict__ B,
    float* __restrict__ C, int lda, int kiters) {
  int lane  = threadIdx.x & 31;
  int wave  = threadIdx.x >> 5;
  int ntile = blockIdx.y * 4 + wave;
  if (ntile >= NT) return;                       // wave-uniform exit; EXEC all-1s for WMMA
  int m0   = blockIdx.x << 4;
  int kgrp = lane >> 4;                          // 0: K 0..7/16..23, 1: K 8..15/24..31
  const __bf16* ap = A + (size_t)(m0 + (lane & 15)) * lda + kgrp * 8;
  const __bf16* bp = B + (size_t)ntile * kiters * 512 + lane * 16;
  v8f c = {};
  for (int kk = 0; kk < kiters; ++kk) {
    __builtin_prefetch(ap + 8 * 32, 0, 0);       // global_prefetch_b8, ~8 k-iters ahead
    v8bf a0 = *(const v8bf*)ap;                  // 16B chunk: K = kk*32 + kgrp*8 + 0..7
    v8bf a1 = *(const v8bf*)(ap + 16);           // 16B chunk: +16
    v16bf a;
#pragma unroll
    for (int i = 0; i < 8; ++i) { a[i] = a0[i]; a[i + 8] = a1[i]; }
    v16bf b = *(const v16bf*)bp;                 // 32B contiguous per lane (pre-packed)
    c = __builtin_amdgcn_wmma_f32_16x16x32_bf16(false, a, false, b, (short)0, c,
                                                false, false);
    ap += 32;
    bp += 512;
  }
  // D layout: lanes 0-15 -> N=lane, M=m0+v; lanes 16-31 -> N=lane-16, M=m0+8+v.
  int n = ntile * 16 + (lane & 15);
  float* cp = C + (size_t)(m0 + kgrp * 8) * NCOLS + n;
#pragma unroll
  for (int v = 0; v < 8; ++v) cp[(size_t)v * NCOLS] = c[v];
}

// ---------------------------------------------------------------------------
__device__ __forceinline__ float sigmoidf_(float x) { return 1.f / (1.f + __expf(-x)); }

// One block per direction (grid=2, block=128). Thread j<120 owns gate row j:
// W_hh row in VGPRs; h and gate vectors in LDS; two split-barriers per step.
__global__ __launch_bounds__(128) void gru_recurrence(
    const float* __restrict__ GI,
    const float* __restrict__ whh_f, const float* __restrict__ whh_b,
    const float* __restrict__ bih_f, const float* __restrict__ bih_b,
    const float* __restrict__ bhh_f, const float* __restrict__ bhh_b,
    float* __restrict__ outF, __bf16* __restrict__ outBf) {
  __shared__ float hsh[HID];
  __shared__ float gis[G3];
  __shared__ float ghs[G3];
  int dir = blockIdx.x;
  int j   = threadIdx.x;
  const float* whh = dir ? whh_b : whh_f;
  const float* bih = dir ? bih_b : bih_f;
  const float* bhh = dir ? bhh_b : bhh_f;
  float wrow[HID];
  float bi = 0.f, bh = 0.f;
  if (j < G3) {
#pragma unroll
    for (int k = 0; k < HID; ++k) wrow[k] = whh[j * HID + k];
    bi = bih[j];
    bh = bhh[j];
  }
  if (j < HID) hsh[j] = 0.f;
  __syncthreads();
  for (int s = 0; s < N_STEPS; ++s) {
    int t = dir ? (N_STEPS - 1 - s) : s;
    const float* gir = GI + (size_t)t * NCOLS + dir * G3;
    if (j < G3) {
      float acc = bh;
#pragma unroll
      for (int k = 0; k < HID; ++k) acc += hsh[k] * wrow[k];
      ghs[j] = acc;
      gis[j] = gir[j] + bi;
    }
    if (j == 127 && s + 1 < N_STEPS) {
      int tn = dir ? (N_STEPS - 2 - s) : (s + 1);
      __builtin_prefetch(GI + (size_t)tn * NCOLS + dir * G3, 0, 0);
    }
    __syncthreads();
    if (j < HID) {
      float r  = sigmoidf_(gis[j] + ghs[j]);
      float z  = sigmoidf_(gis[HID + j] + ghs[HID + j]);
      float nn = tanhf(gis[2 * HID + j] + r * ghs[2 * HID + j]);
      float hn = (1.f - z) * nn + z * hsh[j];
      hsh[j] = hn;
      outF[(size_t)t * (2 * HID) + dir * HID + j] = hn;
      if (outBf) outBf[(size_t)t * K1PAD + dir * HID + j] = (__bf16)hn;
    }
    __syncthreads();
  }
}

// ---------------------------------------------------------------------------
// embs[k,d] = b[k,d] + sum_h emb_in[h] * W[k,h,d];  emb_in = [h_fwd(T-1), h_bwd(0)].
__global__ void heads_kernel(const float* __restrict__ out1,
                             const float* __restrict__ ew,
                             const float* __restrict__ eb,
                             float* __restrict__ out) {
  int tid = threadIdx.x;            // 512 threads: 4 heads x 128 dims
  int k = tid >> 7, d = tid & 127;
  const float* hf = out1 + (size_t)(N_STEPS - 1) * (2 * HID); // fwd @ last step
  const float* hb = out1 + HID;                               // bwd @ step 0
  float acc = eb[k * 128 + d];
#pragma unroll 4
  for (int h = 0; h < HID; ++h) acc += hf[h] * ew[((size_t)k * 2 * HID + h) * 128 + d];
#pragma unroll 4
  for (int h = 0; h < HID; ++h) acc += hb[h] * ew[((size_t)k * 2 * HID + HID + h) * 128 + d];
  out[k * 128 + d] = acc;
}

// ---------------------------------------------------------------------------
extern "C" void kernel_launch(void* const* d_in, const int* in_sizes, int n_in,
                              void* d_out, int out_size, void* d_ws, size_t ws_size,
                              hipStream_t stream) {
  // setup_inputs() insertion order: x, y, then params {l0{w_ih_f,w_hh_f,b_ih_f,
  // b_hh_f,w_ih_b,w_hh_b,b_ih_b,b_hh_b}, l1{...}, emb_w, emb_b}.
  const float* x       = (const float*)d_in[0];
  const float* y       = (const float*)d_in[1];
  const float* w_ih_f0 = (const float*)d_in[2];
  const float* w_hh_f0 = (const float*)d_in[3];
  const float* b_ih_f0 = (const float*)d_in[4];
  const float* b_hh_f0 = (const float*)d_in[5];
  const float* w_ih_b0 = (const float*)d_in[6];
  const float* w_hh_b0 = (const float*)d_in[7];
  const float* b_ih_b0 = (const float*)d_in[8];
  const float* b_hh_b0 = (const float*)d_in[9];
  const float* w_ih_f1 = (const float*)d_in[10];
  const float* w_hh_f1 = (const float*)d_in[11];
  const float* b_ih_f1 = (const float*)d_in[12];
  const float* b_hh_f1 = (const float*)d_in[13];
  const float* w_ih_b1 = (const float*)d_in[14];
  const float* w_hh_b1 = (const float*)d_in[15];
  const float* b_ih_b1 = (const float*)d_in[16];
  const float* b_hh_b1 = (const float*)d_in[17];
  const float* emb_w   = (const float*)d_in[18];
  const float* emb_b   = (const float*)d_in[19];
  float* out = (float*)d_out;

  char* ws = (char*)d_ws;
  __bf16* xcat = (__bf16*)(ws + OFF_XCAT);
  __bf16* bp0  = (__bf16*)(ws + OFF_BP0);
  __bf16* bp1  = (__bf16*)(ws + OFF_BP1);
  float*  gi0  = (float*)(ws + OFF_GI0);
  __bf16* h0bf = (__bf16*)(ws + OFF_H0BF);
  float*  gi1  = (float*)(ws + OFF_GI1);
  float*  out1 = (float*)(ws + OFF_OUT1);
  (void)in_sizes; (void)n_in; (void)out_size; (void)ws_size; (void)WS_NEED;

  // 1) cast/concat X, zero padded bf16 H0, pack both weight sets.
  {
    size_t total = (size_t)N_STEPS * K0PAD;
    build_xcat<<<(unsigned)((total + 255) / 256), 256, 0, stream>>>(x, y, xcat);
  }
  zero_bf16<<<(N_STEPS * K1PAD + 255) / 256, 256, 0, stream>>>(h0bf,
                                                               N_STEPS * K1PAD);
  pack_w<<<(NT * K0ITERS * 512 + 255) / 256, 256, 0, stream>>>(w_ih_f0, w_ih_b0,
                                                               K0, K0ITERS, bp0);
  pack_w<<<(NT * K1ITERS * 512 + 255) / 256, 256, 0, stream>>>(w_ih_f1, w_ih_b1,
                                                               K1, K1ITERS, bp1);

  // 2) layer-0 input projection (the 12.1 GFLOP GEMM) on WMMA.
  dim3 gg(N_STEPS / 16, 4);
  gemm_bf16_wmma<<<gg, 128, 0, stream>>>(xcat, bp0, gi0, K0PAD, K0ITERS);

  // 3) layer-0 recurrence (fwd+bwd concurrently); emits bf16 H0 for GEMM1.
  //    out1 is used as a scratch f32 sink here and overwritten by layer 1.
  gru_recurrence<<<2, 128, 0, stream>>>(gi0, w_hh_f0, w_hh_b0, b_ih_f0, b_ih_b0,
                                        b_hh_f0, b_hh_b0, out1, h0bf);

  // 4) layer-1 input projection (K=96) on WMMA.
  gemm_bf16_wmma<<<gg, 128, 0, stream>>>(h0bf, bp1, gi1, K1PAD, K1ITERS);

  // 5) layer-1 recurrence -> out1 f32.
  gru_recurrence<<<2, 128, 0, stream>>>(gi1, w_hh_f1, w_hh_b1, b_ih_f1, b_ih_b1,
                                        b_hh_f1, b_hh_b1, out1, (__bf16*)nullptr);

  // 6) 4 linear heads -> d_out [4*128].
  heads_kernel<<<1, 512, 0, stream>>>(out1, emb_w, emb_b, out);
}
